// MaSST_13280038879593
// MI455X (gfx1250) — matrix operand order repaired and checked by reading, not compiled, not run on hardware
//
#include <hip/hip_runtime.h>
#include <hip/hip_bf16.h>
#include <math.h>

// ---------------------------------------------------------------------------
// MaSST forward for MI455X (gfx1250, wave32, WMMA).
//  T=128 B=32 D=512 H=512 N=64 A=32, TAU=1
//
//  Parallel (whole GPU, bf16 WMMA):
//    wi_all  = x @ W_ih + bias_ih     (4096 x 1536, K=512)
//    xim_all = x @ W_im               (4096 x  512, K=512)
//    out     = hs @ fc_w + fc_b       (4096 x   32, K=512)
//  Sequential (1 persistent workgroup, 16 wave32, state in LDS):
//    per step: h@W_hm + lu@W_um (+xim) -> tanh -> @fc1 -> argmax(+gumbel)
//              mem row write / row gather, [entry,h]@fc2, 3-gate @W_hh GRU.
// ---------------------------------------------------------------------------

typedef __bf16 bf16_t;
typedef __attribute__((ext_vector_type(16))) __bf16 v16bf;
typedef __attribute__((ext_vector_type(8)))  __bf16 v8bf;
typedef __attribute__((ext_vector_type(8)))  float  v8f;

#define DEVI __device__ __forceinline__

#define T_STEPS 128
#define BSZ 32
#define HH  512
#define NN  64

// A fragment (16x32 bf16, M x K), A row-major with leading dim lda (elements).
// Layout: lanes 0-15 row M=lane, elems 0..15 -> K=k0..k0+15;
// lanes 16-31 same rows, K shifted by +16.
DEVI v16bf load_a_frag(const bf16_t* A, int lda, int m0, int k0, int lane) {
  const bf16_t* p = A + (size_t)(m0 + (lane & 15)) * lda + k0 + ((lane >> 4) << 4);
  v8bf lo = *(const v8bf*)p;
  v8bf hi = *(const v8bf*)(p + 8);
  v16bf r;
#pragma unroll
  for (int i = 0; i < 8; ++i) { r[i] = lo[i]; r[i + 8] = hi[i]; }
  return r;
}

// B fragment (32x16 bf16, K x N) from N-major packed weights Bp[N][K]:
// lane holds column n0+lane%16, K = k0 + (lane/16)*16 + i  -> 32B contiguous.
DEVI v16bf load_b_frag(const bf16_t* Bp, int ldb, int n0, int k0, int lane) {
  const bf16_t* p = Bp + (size_t)(n0 + (lane & 15)) * ldb + k0 + ((lane >> 4) << 4);
  return *(const v16bf*)p;
}

DEVI v8f wmma_bf16(v16bf a, v16bf b, v8f c) {
  return __builtin_amdgcn_wmma_f32_16x16x32_bf16(false, a, false, b, (short)0, c,
                                                 false, false);
}

DEVI float sigmoidf_(float x) { return 1.f / (1.f + expf(-x)); }

// ---------------- generic tiled WMMA GEMM: C = A(bf16,rowmajor) x Bp + bias --
__global__ void wmma_gemm_kernel(const bf16_t* __restrict__ A, int lda,
                                 const bf16_t* __restrict__ Bp, int K,
                                 float* __restrict__ C, int ldc,
                                 const float* __restrict__ bias,
                                 int Mtiles, int Ntiles) {
  const int lane = threadIdx.x & 31;
  const int wave = threadIdx.x >> 5;
  const int wpb  = blockDim.x >> 5;
  const int nT   = Mtiles * Ntiles;
  for (int tIdx = blockIdx.x * wpb + wave; tIdx < nT; tIdx += gridDim.x * wpb) {
    const int mt = tIdx / Ntiles, nt = tIdx % Ntiles;
    v8f acc = {};
    for (int k0 = 0; k0 < K; k0 += 32) {
      v16bf a = load_a_frag(A, lda, mt << 4, k0, lane);
      v16bf b = load_b_frag(Bp, K, nt << 4, k0, lane);
      acc = wmma_bf16(a, b, acc);
    }
    const int n     = (nt << 4) + (lane & 15);
    const float bv  = bias ? bias[n] : 0.f;
    const int mbase = (mt << 4) + ((lane >> 4) << 3);
#pragma unroll
    for (int v = 0; v < 8; ++v)
      C[(size_t)(mbase + v) * ldc + n] = acc[v] + bv;
  }
}

// -------------- weight pack: in f32 [K][N] -> out bf16 [N][K] ---------------
__global__ void pack_bT_kernel(const float* __restrict__ in,
                               bf16_t* __restrict__ out, int K, int N) {
  const size_t total = (size_t)K * N;
  for (size_t i = blockIdx.x * (size_t)blockDim.x + threadIdx.x; i < total;
       i += (size_t)gridDim.x * blockDim.x) {
    const int k = (int)(i / N), n = (int)(i % N);
    out[(size_t)n * K + k] = (bf16_t)in[i];
  }
}

__global__ void cvt_bf_kernel(const float* __restrict__ in,
                              bf16_t* __restrict__ out, size_t total) {
  for (size_t i = blockIdx.x * (size_t)blockDim.x + threadIdx.x; i < total;
       i += (size_t)gridDim.x * blockDim.x)
    out[i] = (bf16_t)in[i];
}

__global__ void zero_f_kernel(float* __restrict__ p, size_t total) {
  for (size_t i = blockIdx.x * (size_t)blockDim.x + threadIdx.x; i < total;
       i += (size_t)gridDim.x * blockDim.x)
    p[i] = 0.f;
}

// ------------------- persistent sequential step kernel ----------------------
__global__ __launch_bounds__(512)
void masst_step_kernel(const float* __restrict__ xim_all,   // [T*B][H]
                       const float* __restrict__ wi_all,    // [T*B][3H]
                       const float* __restrict__ u_noise,   // [T*B][N]
                       const int*   __restrict__ length,    // [B]
                       const float* __restrict__ bias,      // [6H]
                       const float* __restrict__ fc1_b,     // [N]
                       const float* __restrict__ fc2_b,     // [H]
                       const bf16_t* __restrict__ Bp_hm,    // [H][H]
                       const bf16_t* __restrict__ Bp_um,    // [H][N]
                       const bf16_t* __restrict__ Bp_fc1,   // [N][H]
                       const bf16_t* __restrict__ Bp_fc2,   // [H][2H]
                       const bf16_t* __restrict__ Bp_hh,    // [3H][H]
                       float*  __restrict__ mem_ws,         // [B][N][H]
                       bf16_t* __restrict__ hs_bf) {        // [T*B][H]
  extern __shared__ char smem[];
  float*  h_f32  = (float*)smem;                 // 32*512 f32
  float*  hn_f32 = h_f32 + BSZ * HH;             // 32*512 f32
  float*  rh     = hn_f32 + BSZ * HH;            // 32*64  f32
  float*  usage  = rh + BSZ * NN;                // 32*64  f32
  bf16_t* h_bf   = (bf16_t*)(usage + BSZ * NN);  // 32*512 bf16
  bf16_t* hn_bf  = h_bf + BSZ * HH;              // 32*512 bf16
  bf16_t* lu_bf  = hn_bf + BSZ * HH;             // 32*64  bf16
  bf16_t* thca   = lu_bf + BSZ * NN;             // 32*1024 bf16 (tanh / concat)
  int*    amax   = (int*)(thca + BSZ * 2 * HH);  // 32
  int*    lastr  = amax + BSZ;                   // 32

  const int tid  = threadIdx.x;
  const int lane = tid & 31;
  const int wave = tid >> 5;

  // carry init: h = 0, usage = -99999, last_read idx = 0
  for (int i = tid; i < BSZ * HH; i += 512) { h_f32[i] = 0.f; h_bf[i] = (bf16_t)0.f; }
  for (int i = tid; i < BSZ * NN; i += 512) usage[i] = -99999.f;
  if (tid < BSZ) lastr[tid] = 0;
  __syncthreads();

  for (int t = 0; t < T_STEPS; ++t) {
    // phase 0: last_use = sigmoid(usage)
    for (int i = tid; i < BSZ * NN; i += 512)
      lu_bf[i] = (bf16_t)sigmoidf_(usage[i]);
    __syncthreads();

    // phase 1: th = tanh(xim[t] + h @ W_hm + lu @ W_um)   (64 tiles / 16 waves)
#pragma unroll
    for (int it = 0; it < 4; ++it) {
      const int tile = wave + (it << 4);
      const int mt = tile & 1, nt = tile >> 1;
      v8f acc = {};
      for (int k0 = 0; k0 < HH; k0 += 32) {
        v16bf a = load_a_frag(h_bf, HH, mt << 4, k0, lane);
        v16bf b = load_b_frag(Bp_hm, HH, nt << 4, k0, lane);
        acc = wmma_bf16(a, b, acc);
      }
#pragma unroll
      for (int k0 = 0; k0 < NN; k0 += 32) {
        v16bf a = load_a_frag(lu_bf, NN, mt << 4, k0, lane);
        v16bf b = load_b_frag(Bp_um, NN, nt << 4, k0, lane);
        acc = wmma_bf16(a, b, acc);
      }
      const int n     = (nt << 4) + (lane & 15);
      const int mbase = (mt << 4) + ((lane >> 4) << 3);
#pragma unroll
      for (int v = 0; v < 8; ++v) {
        const int m = mbase + v;
        float s = acc[v] + xim_all[((size_t)t * BSZ + m) * HH + n];
        thca[m * HH + n] = (bf16_t)tanhf(s);
      }
    }
    __syncthreads();

    // phase 2: read_head = th @ fc1_w + fc1_b   (8 tiles, waves 0..7)
    if (wave < 8) {
      const int mt = wave & 1, nt = wave >> 1;
      v8f acc = {};
      for (int k0 = 0; k0 < HH; k0 += 32) {
        v16bf a = load_a_frag(thca, HH, mt << 4, k0, lane);
        v16bf b = load_b_frag(Bp_fc1, HH, nt << 4, k0, lane);
        acc = wmma_bf16(a, b, acc);
      }
      const int n     = (nt << 4) + (lane & 15);
      const int mbase = (mt << 4) + ((lane >> 4) << 3);
#pragma unroll
      for (int v = 0; v < 8; ++v)
        rh[(mbase + v) * NN + n] = acc[v] + fc1_b[n];
    }

    // phase 3a: mem[b][w_idx][:] = h[b][:]  (w is one-hot: fill(t) or prev hard)
    {
      const int b     = tid >> 4;
      const int cbase = (tid & 15) << 5;
      const int w_idx = (t < NN) ? t : lastr[b];
      float*       dst = mem_ws + (size_t)((b << 6) + w_idx) * HH + cbase;
      const float* src = h_f32 + b * HH + cbase;
#pragma unroll
      for (int j = 0; j < 32; ++j) dst[j] = src[j];
    }
    __threadfence_block();
    __syncthreads();

    // phase 3b: hard = argmax(read_head + gumbel); usage = (usage-1)*(1-hard)
    if (tid < BSZ) {
      const int b = tid;
      const float* up = u_noise + ((size_t)t * BSZ + b) * NN;
      float best = -1e30f; int bi = 0;
      for (int n = 0; n < NN; ++n) {
        float g = -logf(1e-20f - logf(1e-20f + up[n]));
        float val = rh[b * NN + n] + g;  // TAU == 1
        if (val > best) { best = val; bi = n; }
      }
      for (int n = 0; n < NN; ++n) {
        float uv = usage[b * NN + n] - 1.f;
        usage[b * NN + n] = (n == bi) ? 0.f : uv;
      }
      amax[b] = bi;
      lastr[b] = bi;
    }
    __syncthreads();

    // phase 3c: concat ca = [entry = mem[b][hard], h]  (bf16, stride 1024)
    for (int e = tid; e < BSZ * 2 * HH; e += 512) {
      const int b = e >> 10;
      const int c = e & 1023;
      thca[e] = (c < HH)
                    ? (bf16_t)mem_ws[(size_t)((b << 6) + amax[b]) * HH + c]
                    : h_bf[b * HH + (c - HH)];
    }
    __syncthreads();

    // phase 4: h_new = ca @ fc2_w + fc2_b   (64 tiles, K=1024)
#pragma unroll
    for (int it = 0; it < 4; ++it) {
      const int tile = wave + (it << 4);
      const int mt = tile & 1, nt = tile >> 1;
      v8f acc = {};
      for (int k0 = 0; k0 < 2 * HH; k0 += 32) {
        v16bf a = load_a_frag(thca, 2 * HH, mt << 4, k0, lane);
        v16bf b = load_b_frag(Bp_fc2, 2 * HH, nt << 4, k0, lane);
        acc = wmma_bf16(a, b, acc);
      }
      const int n     = (nt << 4) + (lane & 15);
      const int mbase = (mt << 4) + ((lane >> 4) << 3);
#pragma unroll
      for (int v = 0; v < 8; ++v) {
        const int m = mbase + v;
        float hv = acc[v] + fc2_b[n];
        hn_f32[m * HH + n] = hv;
        hn_bf[m * HH + n]  = (bf16_t)hv;
      }
    }
    __syncthreads();

    // phase 5: fused 3-gate wh = h_new @ W_hh; GRU combine with wi_all[t]
#pragma unroll
    for (int it = 0; it < 4; ++it) {
      const int grp = wave + (it << 4);
      const int mt = grp & 1, nt = grp >> 1;
      v8f accR = {}, accZ = {}, accN = {};
      for (int k0 = 0; k0 < HH; k0 += 32) {
        v16bf a  = load_a_frag(hn_bf, HH, mt << 4, k0, lane);
        v16bf bR = load_b_frag(Bp_hh, HH, (nt << 4),           k0, lane);
        v16bf bZ = load_b_frag(Bp_hh, HH, (nt << 4) + HH,      k0, lane);
        v16bf bN = load_b_frag(Bp_hh, HH, (nt << 4) + 2 * HH,  k0, lane);
        accR = wmma_bf16(a, bR, accR);
        accZ = wmma_bf16(a, bZ, accZ);
        accN = wmma_bf16(a, bN, accN);
      }
      const int n     = (nt << 4) + (lane & 15);
      const int mbase = (mt << 4) + ((lane >> 4) << 3);
      const float bhR = bias[3 * HH + n];
      const float bhZ = bias[3 * HH + n + HH];
      const float bhN = bias[3 * HH + n + 2 * HH];
#pragma unroll
      for (int v = 0; v < 8; ++v) {
        const int m = mbase + v;
        const float* wi = wi_all + ((size_t)t * BSZ + m) * (3 * HH);
        float r  = sigmoidf_(wi[n]          + accR[v] + bhR);
        float z  = sigmoidf_(wi[n + HH]     + accZ[v] + bhZ);
        float nn = tanhf   (wi[n + 2 * HH] + r * (accN[v] + bhN));
        float hnew = hn_f32[m * HH + n];
        float h1 = (1.f - z) * nn + z * hnew;
        float hf = (t < length[m]) ? h1 : h_f32[m * HH + n];
        hs_bf[((size_t)t * BSZ + m) * HH + n] = (bf16_t)hf;
        h_f32[m * HH + n] = hf;
        h_bf[m * HH + n]  = (bf16_t)hf;
      }
    }
    __syncthreads();
  }
}

// ---------------------------------------------------------------------------
extern "C" void kernel_launch(void* const* d_in, const int* in_sizes, int n_in,
                              void* d_out, int out_size, void* d_ws, size_t ws_size,
                              hipStream_t stream) {
  const float* x       = (const float*)d_in[0];
  const int*   length  = (const int*)  d_in[1];
  const float* u_noise = (const float*)d_in[2];
  const float* W_ih    = (const float*)d_in[3];
  const float* W_hh    = (const float*)d_in[4];
  const float* bias    = (const float*)d_in[5];
  const float* W_im    = (const float*)d_in[6];
  const float* W_hm    = (const float*)d_in[7];
  const float* W_um    = (const float*)d_in[8];
  const float* fc1_w   = (const float*)d_in[9];
  const float* fc1_b   = (const float*)d_in[10];
  const float* fc2_w   = (const float*)d_in[11];
  const float* fc2_b   = (const float*)d_in[12];
  const float* fc_w    = (const float*)d_in[13];
  const float* fc_b    = (const float*)d_in[14];
  float* out = (float*)d_out;

  char* ws = (char*)d_ws;
  auto carve = [&](size_t bytes) {
    char* p = ws;
    ws += (bytes + 255) & ~(size_t)255;
    return p;
  };
  const int MR = T_STEPS * BSZ;  // 4096 rows
  bf16_t* Bp_ih  = (bf16_t*)carve((size_t)1536 * 512 * 2);
  bf16_t* Bp_hh  = (bf16_t*)carve((size_t)1536 * 512 * 2);
  bf16_t* Bp_im  = (bf16_t*)carve((size_t)512 * 512 * 2);
  bf16_t* Bp_hm  = (bf16_t*)carve((size_t)512 * 512 * 2);
  bf16_t* Bp_um  = (bf16_t*)carve((size_t)512 * 64 * 2);
  bf16_t* Bp_fc1 = (bf16_t*)carve((size_t)64 * 512 * 2);
  bf16_t* Bp_fc2 = (bf16_t*)carve((size_t)512 * 1024 * 2);
  bf16_t* Bp_fc  = (bf16_t*)carve((size_t)32 * 512 * 2);
  bf16_t* Xbf    = (bf16_t*)carve((size_t)MR * 512 * 2);
  float*  wi_all = (float*) carve((size_t)MR * 1536 * 4);
  float*  xim_all= (float*) carve((size_t)MR * 512 * 4);
  float*  mem_ws = (float*) carve((size_t)BSZ * NN * HH * 4);
  bf16_t* hs_bf  = (bf16_t*)carve((size_t)MR * 512 * 2);

  // weight pack (f32 KxN -> bf16 NxK) + activation convert + mem zero
  pack_bT_kernel<<<1024, 256, 0, stream>>>(W_ih, Bp_ih, 512, 1536);
  pack_bT_kernel<<<1024, 256, 0, stream>>>(W_hh, Bp_hh, 512, 1536);
  pack_bT_kernel<<<512,  256, 0, stream>>>(W_im, Bp_im, 512, 512);
  pack_bT_kernel<<<512,  256, 0, stream>>>(W_hm, Bp_hm, 512, 512);
  pack_bT_kernel<<<128,  256, 0, stream>>>(W_um, Bp_um, 64, 512);
  pack_bT_kernel<<<128,  256, 0, stream>>>(fc1_w, Bp_fc1, 512, 64);
  pack_bT_kernel<<<1024, 256, 0, stream>>>(fc2_w, Bp_fc2, 1024, 512);
  pack_bT_kernel<<<64,   256, 0, stream>>>(fc_w, Bp_fc, 512, 32);
  cvt_bf_kernel<<<1024, 256, 0, stream>>>(x, Xbf, (size_t)MR * 512);
  zero_f_kernel<<<512, 256, 0, stream>>>(mem_ws, (size_t)BSZ * NN * HH);

  // parallel precompute: wi_all = x@W_ih + bias_ih ; xim_all = x@W_im
  wmma_gemm_kernel<<<3072, 256, 0, stream>>>(Xbf, 512, Bp_ih, 512, wi_all, 1536,
                                             bias, MR / 16, 1536 / 16);
  wmma_gemm_kernel<<<1024, 256, 0, stream>>>(Xbf, 512, Bp_im, 512, xim_all, 512,
                                             nullptr, MR / 16, 512 / 16);

  // sequential recurrence: one persistent workgroup, state in LDS
  const size_t smem_bytes =
      (size_t)(2 * BSZ * HH + 2 * BSZ * NN) * 4 +          // f32: h, h_new, rh, usage
      (size_t)(2 * BSZ * HH + BSZ * NN + BSZ * 2 * HH) * 2 + // bf16: h, h_new, lu, th/ca
      (size_t)2 * BSZ * 4;                                 // amax, last_read
  masst_step_kernel<<<1, 512, smem_bytes, stream>>>(
      xim_all, wi_all, u_noise, length, bias, fc1_b, fc2_b, Bp_hm, Bp_um,
      Bp_fc1, Bp_fc2, Bp_hh, mem_ws, hs_bf);

  // final projection: out = hs @ fc_w + fc_b
  wmma_gemm_kernel<<<64, 256, 0, stream>>>(hs_bf, 512, Bp_fc, 512, out, 32,
                                           fc_b, MR / 16, 32 / 16);
}
